// VectorQuantizer_56513179681487
// MI455X (gfx1250) — matrix operand order repaired and checked by reading, not compiled, not run on hardware
//
#include <hip/hip_runtime.h>
#include <hip/hip_bf16.h>
#include <stdint.h>

#define EMBED 256
#define NCODE 1024
#define TILE_M 256            // rows per block (8 waves x 2 rowblocks x 16 rows)
#define DSTEPS (EMBED / 32)   // 8 WMMA reduction steps of K=32
#define KTILE 16
#define KCHUNK 128            // codes per LDS-staged E chunk (128 KB interleaved hi/lo)
#define NCHUNK (NCODE / KCHUNK)
#define BUF_U32 32768         // 128 KB buffer in u32 units

typedef __attribute__((ext_vector_type(16))) __bf16 v16bf;
typedef __attribute__((ext_vector_type(8)))  float  v8f;
typedef unsigned int u32x4 __attribute__((ext_vector_type(4)));
typedef int          i32x8 __attribute__((ext_vector_type(8)));
typedef int          i32x4 __attribute__((ext_vector_type(4)));

union Frag {
    uint32_t u[8];
    uint4    q[2];
    v16bf    v;
};

// fp32 -> bf16 hi/lo split with round-to-nearest-even, as raw u16 payloads.
__device__ __forceinline__ void bf16_split(float x, uint16_t& hi, uint16_t& lo) {
    uint32_t u = __float_as_uint(x);
    uint32_t r = (u + 0x7FFFu + ((u >> 16) & 1u)) & 0xFFFF0000u;
    hi = (uint16_t)(r >> 16);
    float res = x - __uint_as_float(r);
    uint32_t u2 = __float_as_uint(res);
    uint32_t r2 = (u2 + 0x7FFFu + ((u2 >> 16) & 1u)) & 0xFFFF0000u;
    lo = (uint16_t)(r2 >> 16);
}

// One TDM descriptor: 1-D contiguous 128 KB chunk (16384 x 8B) -> LDS.
__device__ __forceinline__ void tdm_load_chunk(const uint16_t* gsrc, uint32_t ldsByteAddr) {
    uint64_t ga = (uint64_t)(uintptr_t)gsrc;
    u32x4 g0;
    g0.x = 1u;                                    // count=1, user mode
    g0.y = ldsByteAddr;                           // lds_addr
    g0.z = (uint32_t)(ga & 0xFFFFFFFFu);          // global_addr[31:0]
    g0.w = (uint32_t)((ga >> 32) & 0x01FFFFFFu)   // global_addr[56:32]
         | (2u << 30);                            // type=2 (image)
    i32x8 g1;
    g1[0] = 0x00030000;   // workgroup_mask=0, data_size=3 (8B)
    g1[1] = 0x40000000;   // tensor_dim0 = 16384 (lo16 at [31:16])
    g1[2] = 0x00010000;   // tensor_dim0 hi=0, tensor_dim1 = 1
    g1[3] = 0x40000000;   // tensor_dim1 hi=0, tile_dim0 = 16384
    g1[4] = 0x00000000;   // tile_dim1 = 0 (1-D), tile_dim2 = 0
    g1[5] = 0x00004000;   // tensor_dim0_stride = 16384
    g1[6] = 0;
    g1[7] = 0;
    i32x4 g2 = {0, 0, 0, 0};     // 1-D tile: groups 2/3 unused
    i32x4 g3 = {0, 0, 0, 0};
    i32x8 g4 = {0, 0, 0, 0, 0, 0, 0, 0};
    __builtin_amdgcn_tensor_load_to_lds(g0, g1, g2, g3, g4, 0);
}

// Transpose E [256][1024] -> per-code interleaved hi/lo bf16 [1024][512]
// (EtI[k][0:256]=hi, EtI[k][256:512]=lo) plus fp32 transposed copy.
__global__ __launch_bounds__(256) void vq_prep_embed(
    const float* __restrict__ E,
    uint16_t* __restrict__ EtI,
    float* __restrict__ EtF)
{
    int gid = blockIdx.x * 256 + threadIdx.x;   // 0 .. 262143
    int k = gid & (NCODE - 1);
    int d = gid >> 10;
    float v = E[(size_t)d * NCODE + k];
    uint16_t h, l;
    bf16_split(v, h, l);
    EtI[(size_t)k * 512 + d]       = h;
    EtI[(size_t)k * 512 + 256 + d] = l;
    EtF[(size_t)k * EMBED + d]     = v;
}

// ||e_k||^2 per code.
__global__ __launch_bounds__(256) void vq_prep_norm(
    const float* __restrict__ E, float* __restrict__ en2)
{
    int k = blockIdx.x * 256 + threadIdx.x;     // grid = 4 -> 1024 codes
    float s = 0.f;
    #pragma unroll 8
    for (int d = 0; d < EMBED; ++d) {
        float v = E[(size_t)d * NCODE + k];
        s += v * v;
    }
    en2[k] = s;
}

// Core: WMMA bf16 hi/lo split GEMM S = X*Et^T, fused argmax of (S - 0.5*||e||^2).
// E chunks staged to LDS by the Tensor Data Mover, double-buffered.
// Each wave owns 2 rowblocks (32 rows): two independent accumulator chains
// fill the WMMA->VALU WAR hazard slots and halve LDS B-fragment traffic.
__global__ __launch_bounds__(256) void vq_argmin(
    const float* __restrict__ X,
    const uint16_t* __restrict__ EtI,       // [1024][512] interleaved hi/lo
    const float* __restrict__ en2,
    int* __restrict__ idxOut)
{
    // 256 KB: buf0 | buf1. X staging borrows buf0; E chunks alternate buf1, buf0, ...
    __shared__ uint32_t lds[2 * BUF_U32];

    const int tid = threadIdx.x;
    const int wave = tid >> 5;
    const int lane = tid & 31;
    const int rowBase = blockIdx.x * TILE_M;

    // Kick off TDM for chunk 0 into buf1 before X staging (overlap copy w/ convert).
    if (wave == 0)
        tdm_load_chunk(EtI, (uint32_t)(uintptr_t)&lds[BUF_U32]);

    const int col  = lane & 15;
    const int half = lane >> 4;

    Frag aHi[2][DSTEPS], aLo[2][DSTEPS];

    // Stage X in two 128-row phases through buf0; waves grab their A frags
    // (K-loop invariant) from the phase that holds their rows.
    #pragma unroll
    for (int phase = 0; phase < 2; ++phase) {
        for (int i = 0; i < 64; ++i) {
            int p   = tid + i * 256;          // pair index, 0..16383
            int row = p >> 7;                 // local row 0..127
            int dp  = (p & 127) << 1;         // even d
            float2 xv = *(const float2*)(X +
                (size_t)(rowBase + phase * 128 + row) * EMBED + dp);
            uint16_t h0, l0, h1, l1;
            bf16_split(xv.x, h0, l0);
            bf16_split(xv.y, h1, l1);
            int rowblk = row >> 4, rl = row & 15;
            int dstep  = dp >> 5,  dm = dp & 31;
            // ISA 16-bit A 16x32: lanes 0-15 hold K{0-7,16-23}, lanes 16-31 K{8-15,24-31}
            int ln   = rl + (((dm >> 3) & 1) << 4);
            int elem = (dm & 7) + ((dm >> 4) << 3);     // even (dp even)
            int base = (((rowblk * DSTEPS + dstep) * 32 + ln) << 3) + (elem >> 1);
            lds[base]         = (uint32_t)h0 | ((uint32_t)h1 << 16);
            lds[base + 16384] = (uint32_t)l0 | ((uint32_t)l1 << 16);
        }
        __syncthreads();
        // Waves 0-3 own rowblocks 0-7 (phase 0); waves 4-7 own 8-15 (phase 1).
        if ((wave >> 2) == phase) {
            #pragma unroll
            for (int j = 0; j < 2; ++j) {
                int localRb = ((2 * wave + j) & 7);
                #pragma unroll
                for (int s = 0; s < DSTEPS; ++s) {
                    int base = ((localRb * DSTEPS + s) * 32 + lane) << 3;
                    aHi[j][s].q[0] = *(const uint4*)&lds[base];
                    aHi[j][s].q[1] = *(const uint4*)&lds[base + 4];
                    aLo[j][s].q[0] = *(const uint4*)&lds[base + 16384];
                    aLo[j][s].q[1] = *(const uint4*)&lds[base + 16384 + 4];
                }
            }
        }
        __syncthreads();   // frag reads done before buf0 is re-filled
    }

    float best[2][8];
    int   bidx[2][8];
    #pragma unroll
    for (int j = 0; j < 2; ++j)
        #pragma unroll
        for (int r = 0; r < 8; ++r) { best[j][r] = -3.4e38f; bidx[j][r] = 0; }

    for (int c = 0; c < NCHUNK; ++c) {
        // chunk c lives in buf[(c+1)&1]; prefetch chunk c+1 into buf[c&1].
        if (wave == 0) {
            if (c + 1 < NCHUNK) {
                tdm_load_chunk(EtI + (size_t)(c + 1) * KCHUNK * 512,
                               (uint32_t)(uintptr_t)&lds[(c & 1) * BUF_U32]);
                __builtin_amdgcn_s_wait_tensorcnt(1);   // chunk c complete (in-order)
            } else {
                __builtin_amdgcn_s_wait_tensorcnt(0);
            }
        }
        __syncthreads();    // release all waves: chunk c visible in LDS

        const uint32_t* buf = &lds[((c + 1) & 1) * BUF_U32];

        for (int kt = 0; kt < KCHUNK; kt += KTILE) {
            v8f acc0 = {0.f, 0.f, 0.f, 0.f, 0.f, 0.f, 0.f, 0.f};
            v8f acc1 = {0.f, 0.f, 0.f, 0.f, 0.f, 0.f, 0.f, 0.f};
            // one base pointer; all 4 loads per dstep are immediate offsets
            const uint32_t* bp = buf + (kt + col) * 256 + half * 4;
            #pragma unroll
            for (int s = 0; s < DSTEPS; ++s) {
                Frag bHi, bLo;      // code block: 256 u32 hi then... hi[0:128], lo[128:256]
                bHi.q[0] = *(const uint4*)&bp[s * 16];
                bHi.q[1] = *(const uint4*)&bp[s * 16 + 8];
                bLo.q[0] = *(const uint4*)&bp[s * 16 + 128];
                bLo.q[1] = *(const uint4*)&bp[s * 16 + 136];
                acc0 = __builtin_amdgcn_wmma_f32_16x16x32_bf16(
                    false, aHi[0][s].v, false, bHi.v, (short)0, acc0, false, false);
                acc1 = __builtin_amdgcn_wmma_f32_16x16x32_bf16(
                    false, aHi[1][s].v, false, bHi.v, (short)0, acc1, false, false);
                acc0 = __builtin_amdgcn_wmma_f32_16x16x32_bf16(
                    false, aHi[0][s].v, false, bLo.v, (short)0, acc0, false, false);
                acc1 = __builtin_amdgcn_wmma_f32_16x16x32_bf16(
                    false, aHi[1][s].v, false, bLo.v, (short)0, acc1, false, false);
                acc0 = __builtin_amdgcn_wmma_f32_16x16x32_bf16(
                    false, aLo[0][s].v, false, bHi.v, (short)0, acc0, false, false);
                acc1 = __builtin_amdgcn_wmma_f32_16x16x32_bf16(
                    false, aLo[1][s].v, false, bHi.v, (short)0, acc1, false, false);
            }
            int   k  = c * KCHUNK + kt + col;
            float eb = 0.5f * en2[k];
            #pragma unroll
            for (int r = 0; r < 8; ++r) {
                float s0 = acc0[r] - eb;   // = x.e - 0.5||e||^2 ; argmax == argmin dist
                float s1 = acc1[r] - eb;
                if (s0 > best[0][r]) { best[0][r] = s0; bidx[0][r] = k; }
                if (s1 > best[1][r]) { best[1][r] = s1; bidx[1][r] = k; }
            }
        }
        __syncthreads();    // all waves done reading buf[(c+1)&1] before re-fill
    }

    // Butterfly argmax over the 16 lanes of each half; smaller index wins ties.
    #pragma unroll
    for (int off = 1; off < 16; off <<= 1) {
        #pragma unroll
        for (int j = 0; j < 2; ++j)
            #pragma unroll
            for (int r = 0; r < 8; ++r) {
                float os = __shfl_xor(best[j][r], off, 32);
                int   oi = __shfl_xor(bidx[j][r], off, 32);
                if (os > best[j][r] || (os == best[j][r] && oi < bidx[j][r])) {
                    best[j][r] = os; bidx[j][r] = oi;
                }
            }
    }
    if (col == 0) {
        #pragma unroll
        for (int j = 0; j < 2; ++j) {
            int rbase = rowBase + (2 * wave + j) * 16 + half * 8;   // rows r + 8*half
            #pragma unroll
            for (int r = 0; r < 8; ++r) idxOut[rbase + r] = bidx[j][r];
        }
    }
}

// Gather + emit quantized_st, codes=[x|q], float(idx).
__global__ __launch_bounds__(256) void vq_emit(
    const float* __restrict__ X, const float* __restrict__ EtF,
    const int* __restrict__ idx,
    float* __restrict__ outQ, float* __restrict__ outC, float* __restrict__ outI,
    int nRows)
{
    int t   = blockIdx.x * 256 + threadIdx.x;
    int row = t >> 6;                 // 64 threads x float4 = 256 floats/row
    int c   = (t & 63) << 2;
    if (row >= nRows) return;
    int k = idx[row];
    float4 q = *(const float4*)(EtF + (size_t)k   * EMBED + c);
    float4 x = *(const float4*)(X   + (size_t)row * EMBED + c);
    float4 qst;   // exact emulation of x + (q - x) in fp32
    qst.x = x.x + (q.x - x.x);
    qst.y = x.y + (q.y - x.y);
    qst.z = x.z + (q.z - x.z);
    qst.w = x.w + (q.w - x.w);
    *(float4*)(outQ + (size_t)row * EMBED + c) = qst;
    *(float4*)(outC + (size_t)row * (2 * EMBED) + c) = x;
    *(float4*)(outC + (size_t)row * (2 * EMBED) + EMBED + c) = q;
    if ((t & 63) == 0) outI[row] = (float)k;
}

extern "C" void kernel_launch(void* const* d_in, const int* in_sizes, int n_in,
                              void* d_out, int out_size, void* d_ws, size_t ws_size,
                              hipStream_t stream) {
    const float* X = (const float*)d_in[0];   // [16,4096,256] fp32
    const float* E = (const float*)d_in[1];   // [256,1024] fp32
    const int N = in_sizes[0] / EMBED;        // 65536 rows

    char* ws = (char*)d_ws;
    uint16_t* EtI = (uint16_t*)ws;   ws += (size_t)NCODE * 512 * 2;     // 1 MB interleaved hi/lo
    float*    EtF = (float*)ws;      ws += (size_t)NCODE * EMBED * 4;   // 1 MB
    float*    en2 = (float*)ws;      ws += (size_t)NCODE * 4;
    int*      idx = (int*)ws;        ws += (size_t)N * 4;

    float* outQ = (float*)d_out;                    // [N,256]
    float* outC = outQ + (size_t)N * EMBED;         // [N,512]
    float* outI = outC + (size_t)N * 2 * EMBED;     // [N]

    vq_prep_embed<<<(NCODE * EMBED) / 256, 256, 0, stream>>>(E, EtI, EtF);
    vq_prep_norm<<<NCODE / 256, 256, 0, stream>>>(E, en2);
    vq_argmin<<<N / TILE_M, 256, 0, stream>>>(X, EtI, en2, idx);
    vq_emit<<<(N * 64) / 256, 256, 0, stream>>>(X, EtF, idx, outQ, outC, outI, N);
}
